// Conv2D_Manual_56882546868537
// MI455X (gfx1250) — matrix operand order repaired and checked by reading, not compiled, not run on hardware
//
#include <hip/hip_runtime.h>

typedef __attribute__((ext_vector_type(16))) __bf16        v16bf;
typedef __attribute__((ext_vector_type(8)))  float         v8f;
typedef __attribute__((ext_vector_type(8)))  unsigned int  v8u;
typedef __attribute__((ext_vector_type(4)))  unsigned int  v4u;
typedef __attribute__((ext_vector_type(8)))  int           v8i;
typedef __attribute__((ext_vector_type(4)))  int           v4i;

// ---------------------------------------------------------------------------
// Conv 3x3 valid: B=32, C=64, K=128, H=W=130 -> out [32,128,128,128].
// Implicit GEMM: M=k(128), N=j(64/block), reduction=C*9=576 (18 chunks of 32).
// x slab (64ch x 3rows x 66cols fp32) staged to LDS via the Tensor Data Mover.
// ---------------------------------------------------------------------------
#define RED      576      // C*3*3
#define NCHUNK   18       // 576 / 32
#define XROW     66       // slab row length (fp32, unpadded -> TDM contiguous)
#define XCH      198      // 3 * XROW, per-channel stride in slab
#define IMSTR    34       // padded im2col kk stride (32 used, keeps u32 align)

#if defined(__has_builtin)
#if __has_builtin(__builtin_amdgcn_tensor_load_to_lds) && \
    __has_builtin(__builtin_amdgcn_s_wait_tensorcnt)
#define HAVE_TDM 1
#endif
#endif
#ifndef HAVE_TDM
#define HAVE_TDM 0
#endif

__device__ __forceinline__ unsigned short f32_to_bf16_rne(float f) {
  union { float f; unsigned int u; } x; x.f = f;
  unsigned int u = x.u;
  unsigned int r = u + 0x7FFFu + ((u >> 16) & 1u);   // round-to-nearest-even
  return (unsigned short)(r >> 16);
}

// Convert weights [K=128][C*3*3=576] fp32 -> bf16 workspace, same layout.
// Inner index is exactly the GEMM reduction index kk = c*9 + u*3 + v.
__global__ void Conv2D_wconvert_kernel(const float* __restrict__ w,
                                       unsigned short* __restrict__ bt,
                                       int n) {
  int i = blockIdx.x * blockDim.x + threadIdx.x;
  if (i < n) bt[i] = f32_to_bf16_rne(w[i]);
}

__global__ __launch_bounds__(256)
void Conv2D_wmma_kernel(const float* __restrict__ x,
                        const unsigned short* __restrict__ bt,   // [128][576] bf16
                        const float* __restrict__ bias,          // [128]
                        float* __restrict__ out) {
  __shared__ float          xs[64 * XCH];    // x slab: 64ch x 3rows x 66cols fp32
  __shared__ unsigned short im[64 * IMSTR];  // im2col chunk: 64 j x 32(34) kk bf16

  const int j0   = blockIdx.x * 64;   // spatial-j tile base (0 or 64)
  const int i    = blockIdx.y;        // output row (0..127)
  const int b    = blockIdx.z;        // batch
  const int tid  = threadIdx.x;
  const int lane = tid & 31;
  const int wave = tid >> 5;          // 8 waves: one per 16 output channels
  const int m    = lane & 15;
  const int hi   = lane >> 4;

  // ---- stage x slab (fp32) into LDS ---------------------------------------
#if HAVE_TDM
  if (wave == 0) {
    // D# descriptor (ISA 8.3-8.6): 3D tile 66 x 3 x 64 of 4-byte elements,
    // strides 130 (rows) and 16900 (channels); no padding, no OOB clipping.
    unsigned long long ga =
        (unsigned long long)(const void*)(x + (long)b * 64 * 16900 +
                                          (long)i * 130 + j0);
    unsigned lds = (unsigned)(unsigned long long)(void*)xs;

    v4u g0;
    g0[0] = 1u;                                   // count=1 (valid descriptor)
    g0[1] = lds;                                  // lds_addr
    g0[2] = (unsigned)(ga & 0xFFFFFFFFu);         // global_addr[31:0]
    g0[3] = (unsigned)((ga >> 32) & 0x1FFFFFFu) | 0x80000000u; // [56:32]|type=2

    const unsigned bigdim = 1u << 30;             // huge tensor dims: no clip
    v8i g1;
    g1[0] = 0x20000;                              // data_size=2 -> 4 bytes
    g1[1] = (int)((bigdim & 0xFFFFu) << 16);      // tensor_dim0[15:0]
    g1[2] = (int)((bigdim >> 16) | ((bigdim & 0xFFFFu) << 16)); // d0 hi | d1 lo
    g1[3] = (int)((bigdim >> 16) | (66u << 16));  // d1 hi | tile_dim0=66
    g1[4] = (int)(3u | (64u << 16));              // tile_dim1=3 | tile_dim2=64
    g1[5] = 130;                                  // tensor_dim0_stride = W
    g1[6] = (int)((16900u & 0xFFFFu) << 16);      // dim1_stride[15:0] = H*W
    g1[7] = (int)(16900u >> 16);                  // dim1_stride[47:16]

    v4i g2;
    g2[0] = (int)bigdim;                          // tensor_dim2 (no clip)
    g2[1] = 0; g2[2] = 0; g2[3] = 0;              // dim3 / stride2 / tile_dim3
    v4i g3 = {0, 0, 0, 0};                        // dim3_stride/dim4/tile_dim4
    v8i g4 = {0, 0, 0, 0, 0, 0, 0, 0};            // unused trailing group (VADDR4)

    __builtin_amdgcn_tensor_load_to_lds(g0, g1, g2, g3, g4, 0);
    __builtin_amdgcn_s_wait_tensorcnt(0);
  }
  // loop-top __syncthreads() publishes the slab to all waves
#else
  const float* xb = x + (long)b * 64 * 16900;
  for (int idx = tid; idx < 64 * 3 * XROW; idx += 256) {
    int c   = idx / XCH;
    int rem = idx - c * XCH;
    int u   = rem / XROW;
    int jj  = rem - u * XROW;
    xs[c * XCH + u * XROW + jj] = xb[(c * 130 + (i + u)) * 130 + (j0 + jj)];
  }
#endif

  // ---- accumulators: 4 tiles of 16x16 (k-slice x 64 j) ---------------------
  v8f acc0 = {}, acc1 = {}, acc2 = {}, acc3 = {};

  const int kbase = wave * 16;
  const unsigned int* btw = (const unsigned int*)(bt + (kbase + m) * RED);
  const unsigned int* imw = (const unsigned int*)im;

  for (int t = 0; t < NCHUNK; ++t) {
    const int kk0 = t * 32;
    __syncthreads();                       // slab ready / prev chunk consumed
    // build im2col chunk im[j][kk] from the fp32 slab (convert to bf16 here)
    for (int e = tid; e < 64 * 32; e += 256) {
      int jl  = e >> 5;
      int kkl = e & 31;
      int kk  = kk0 + kkl;
      int c   = kk / 9;
      int r9  = kk - c * 9;
      int u   = r9 / 3;
      int v2  = r9 - u * 3;
      im[jl * IMSTR + kkl] =
          f32_to_bf16_rne(xs[c * XCH + u * XROW + jl + v2]);
    }
    __syncthreads();

    // A fragment (16x32 bf16): lanes 0-15: K 0-7,16-23; lanes 16-31: K 8-15,24-31
    v8u au;
#pragma unroll
    for (int v = 0; v < 8; ++v) {
      int kk = (v < 4 ? 2 * v : 16 + 2 * (v - 4)) + (hi ? 8 : 0) + kk0;
      au[v] = btw[kk >> 1];
    }
    union { v8u u; v16bf h; } A; A.u = au;

    // B fragments (32x16 bf16): lanes 0-15: K 0-7,16-23; lanes 16-31: K 8-15,24-31
#pragma unroll
    for (int st = 0; st < 4; ++st) {
      v8u bu;
#pragma unroll
      for (int v = 0; v < 8; ++v) {
        int kkl = (v < 4 ? 2 * v : 16 + 2 * (v - 4)) + (hi ? 8 : 0);
        int j   = st * 16 + m;
        bu[v] = imw[(j * IMSTR + kkl) >> 1];
      }
      union { v8u u; v16bf h; } Bf; Bf.u = bu;
      v8f& acc = (st == 0 ? acc0 : st == 1 ? acc1 : st == 2 ? acc2 : acc3);
      acc = __builtin_amdgcn_wmma_f32_16x16x32_bf16(
          false, A.h, false, Bf.h, (short)0, acc, false, false);
    }
  }

  // ---- store: C/D layout — N=j across lanes, M=k = r + 8*hi ----------------
#pragma unroll
  for (int st = 0; st < 4; ++st) {
    v8f acc = (st == 0 ? acc0 : st == 1 ? acc1 : st == 2 ? acc2 : acc3);
    int j = j0 + st * 16 + m;
    float bv = bias[j];   // reference quirk: bias broadcasts over LAST axis (j)
#pragma unroll
    for (int r = 0; r < 8; ++r) {
      int k = kbase + r + 8 * hi;
      out[(((long)b * 128 + k) * 128 + i) * 128 + j] = acc[r] + bv;
    }
  }
}

extern "C" void kernel_launch(void* const* d_in, const int* in_sizes, int n_in,
                              void* d_out, int out_size, void* d_ws, size_t ws_size,
                              hipStream_t stream) {
  const float* x      = (const float*)d_in[0];   // [32,64,130,130]
  const float* weight = (const float*)d_in[1];   // [128,64,3,3]
  const float* bias   = (const float*)d_in[2];   // [128]
  float* out          = (float*)d_out;           // [32,128,128,128]

  unsigned short* bt = (unsigned short*)d_ws;    // bf16 weights [128][576] (144 KB)

  const int nw = 128 * RED;                      // 73728 elements
  Conv2D_wconvert_kernel<<<(nw + 255) / 256, 256, 0, stream>>>(weight, bt, nw);

  dim3 grid(128 / 64, 128, 32);                  // (j-tile, i, b)
  Conv2D_wmma_kernel<<<grid, 256, 0, stream>>>(x, bt, bias, out);
}
